// QuanvolutionClassifier_65481071405526
// MI455X (gfx1250) — matrix operand order repaired and checked by reading, not compiled, not run on hardware
//
#include <hip/hip_runtime.h>
#include <hip/hip_bf16.h>

typedef __attribute__((ext_vector_type(16))) _Float16 v16h;
typedef __attribute__((ext_vector_type(8)))  float    v8f;

#define NCHUNK 49                    // K = 1568 = 49 * 32  (784 pixels x {cos,sin})
#define W2_ELEMS (NCHUNK * 32 * 16)  // packed B-matrix halves

struct C2 { float re, im; };
__device__ inline C2 cmul(C2 a, C2 b){ return {a.re*b.re - a.im*b.im, a.re*b.im + a.im*b.re}; }
__device__ inline C2 cadd(C2 a, C2 b){ return {a.re + b.re, a.im + b.im}; }
__device__ inline float cabs2(C2 a){ return a.re*a.re + a.im*a.im; }
__device__ inline float redotc(C2 a, C2 b){ return a.re*b.re + a.im*b.im; } // Re(a*conj(b))

// ---------------------------------------------------------------------------
// Kernel 1: collapse the depth-4 single-qubit circuit per wire into
//   z(v) = alpha_k + beta_k*cos(v) + gamma_k*sin(v),
// bake permutation + coefficients into a WMMA-B-layout weight buffer, and
// fold alpha into the bias.
// ---------------------------------------------------------------------------
__global__ void quanv_precompute(const float* __restrict__ params,  // (4,4,3)
                                 const float* __restrict__ w,       // (10,784)
                                 const float* __restrict__ bias,    // (10)
                                 _Float16* __restrict__ w2,         // [NCHUNK][32][16]
                                 float* __restrict__ biasp)         // [16]
{
    __shared__ float s_alpha[4], s_beta[4], s_gamma[4];
    const int t = threadIdx.x;

    if (t < 4) {
        // U = M3*M2*M1*M0, with M = Rz(rz2)*Ry(ry)*Rz(rz1)
        C2 u00 = {1.f, 0.f}, u01 = {0.f, 0.f}, u10 = {0.f, 0.f}, u11 = {1.f, 0.f};
        for (int d = 0; d < 4; ++d) {
            float rz1 = params[(d*4 + t)*3 + 0];
            float ry  = params[(d*4 + t)*3 + 1];
            float rz2 = params[(d*4 + t)*3 + 2];
            C2 p1  = { __cosf(0.5f*rz1), -__sinf(0.5f*rz1) };
            C2 p1c = { p1.re, -p1.im };
            C2 p2  = { __cosf(0.5f*rz2), -__sinf(0.5f*rz2) };
            C2 p2c = { p2.re, -p2.im };
            float c = __cosf(0.5f*ry), s = __sinf(0.5f*ry);
            C2 m00 = cmul(p2,  { c*p1.re,  c*p1.im });
            C2 m01 = cmul(p2,  {-s*p1c.re, -s*p1c.im});
            C2 m10 = cmul(p2c, { s*p1.re,  s*p1.im });
            C2 m11 = cmul(p2c, { c*p1c.re, c*p1c.im});
            C2 n00 = cadd(cmul(m00, u00), cmul(m01, u10));
            C2 n01 = cadd(cmul(m00, u01), cmul(m01, u11));
            C2 n10 = cadd(cmul(m10, u00), cmul(m11, u10));
            C2 n11 = cadd(cmul(m10, u01), cmul(m11, u11));
            u00 = n00; u01 = n01; u10 = n10; u11 = n11;
        }
        float A  = cabs2(u00) - cabs2(u10);   // coeff of cos^2(v/2)
        float Bv = cabs2(u01) - cabs2(u11);   // coeff of sin^2(v/2)
        s_alpha[t] = 0.5f * (A + Bv);
        s_beta[t]  = 0.5f * (A - Bv);
        s_gamma[t] = redotc(u00, u01) - redotc(u10, u11);
    }
    __syncthreads();

    // bias'[o] = bias[o] + sum_f alpha_{f&3} * w[o,f]   (feature f has wire = f&3)
    if (t < 16) {
        float acc = 0.f;
        if (t < 10) {
            acc = bias[t];
            for (int f = 0; f < 784; ++f)
                acc += s_alpha[f & 3] * w[t*784 + f];
        }
        biasp[t] = acc;
    }

    // Pack weights into B lanes in this code's contiguous k order (A uses the same):
    // chunk c, lane l, halves j=0..15:  n = l&15,  K = c*32 + (l>=16)*16 + j
    // K even -> cos weight (beta), K odd -> sin weight (gamma); pixel p = K>>1.
    for (int idx = t; idx < W2_ELEMS; idx += blockDim.x) {
        int c = idx >> 9;         // / (32*16)
        int r = idx & 511;
        int l = r >> 4;           // lane
        int j = r & 15;           // element within lane
        int n = l & 15;           // output column
        int K = c*32 + ((l >> 4) << 4) + j;
        int p = K >> 1;           // pixel index in row-major 28x28
        int trig = K & 1;         // 0=cos, 1=sin
        float val = 0.f;
        if (n < 10) {
            int row = p / 28, col = p - row*28;
            int wire = ((row & 1) << 1) | (col & 1);
            int f = 4*(14*(row >> 1) + (col >> 1)) + wire;  // patch permutation
            float coef = trig ? s_gamma[wire] : s_beta[wire];
            val = coef * w[n*784 + f];
        }
        w2[idx] = (_Float16)val;
    }
}

// ---------------------------------------------------------------------------
// Kernel 2: streaming sin/cos + WMMA GEMM + LDS-transpose log-softmax.
// One wave per 16-row tile; 49 x v_wmma_f32_16x16x32_f16 per tile.
// Epilogue: lanes dump the C tile column-major into LDS (2x ds_store_b128,
// row-pad 20 => 16B aligned, conflict-free), then lanes 0..15 each own one
// batch row and do the 10-class log-softmax serially in registers.
// ---------------------------------------------------------------------------
__global__ void __launch_bounds__(256) quanv_main(const float* __restrict__ x,
                                                  const _Float16* __restrict__ w2,
                                                  const float* __restrict__ biasp,
                                                  float* __restrict__ out)
{
    __shared__ __align__(16) float ldsT[8][16][20];   // [wave][col n][row m], pad 20

    const int lane = threadIdx.x & 31;
    const int wave = threadIdx.x >> 5;
    const int tile = blockIdx.x * 8 + wave;       // 64 blocks * 8 waves = 512 tiles
    const int m    = lane & 15;                    // row within tile (A) / column N (B,C,D)
    const int hs   = lane >> 4;                    // lane-group select

    const float* rowptr = x + (size_t)(tile*16 + m) * 784;

    v8f acc = {};
    for (int c = 0; c < NCHUNK; ++c) {
        const float4 g0 = *(const float4*)(rowptr + c*16 + hs*8);
        const float4 g1 = *(const float4*)(rowptr + c*16 + hs*8 + 4);
        v16h a;
        a[0]  = (_Float16)__cosf(g0.x);  a[1]  = (_Float16)__sinf(g0.x);
        a[2]  = (_Float16)__cosf(g0.y);  a[3]  = (_Float16)__sinf(g0.y);
        a[4]  = (_Float16)__cosf(g0.z);  a[5]  = (_Float16)__sinf(g0.z);
        a[6]  = (_Float16)__cosf(g0.w);  a[7]  = (_Float16)__sinf(g0.w);
        a[8]  = (_Float16)__cosf(g1.x);  a[9]  = (_Float16)__sinf(g1.x);
        a[10] = (_Float16)__cosf(g1.y);  a[11] = (_Float16)__sinf(g1.y);
        a[12] = (_Float16)__cosf(g1.z);  a[13] = (_Float16)__sinf(g1.z);
        a[14] = (_Float16)__cosf(g1.w);  a[15] = (_Float16)__sinf(g1.w);
        v16h b = *(const v16h*)(w2 + (size_t)(c*32 + lane) * 16);
        acc = __builtin_amdgcn_wmma_f32_16x16x32_f16(
                  false, a, false, b, (short)0, acc, false, false);
    }

    // C/D layout: lane = column n (mod 16), acc[v] = row v + 8*hs.
    // Transpose through LDS: column-major store, two 16B writes per lane.
    float* colp = &ldsT[wave][m][8*hs];
    *(float4*)(colp + 0) = make_float4(acc[0], acc[1], acc[2], acc[3]);
    *(float4*)(colp + 4) = make_float4(acc[4], acc[5], acc[6], acc[7]);
    __syncthreads();

    if (lane < 16) {
        const int row = lane;                     // batch row within tile
        float v[10];
        float mx = -__builtin_inff();
#pragma unroll
        for (int n = 0; n < 10; ++n) {
            v[n] = ldsT[wave][n][row] + biasp[n];
            mx = fmaxf(mx, v[n]);
        }
        float s = 0.f;
#pragma unroll
        for (int n = 0; n < 10; ++n)
            s += __expf(v[n] - mx);
        const float lg = mx + __logf(s);
        float* orow = out + (size_t)(tile*16 + row) * 10;
#pragma unroll
        for (int n = 0; n < 10; ++n)
            orow[n] = v[n] - lg;
    }
}

extern "C" void kernel_launch(void* const* d_in, const int* in_sizes, int n_in,
                              void* d_out, int out_size, void* d_ws, size_t ws_size,
                              hipStream_t stream) {
    const float* x      = (const float*)d_in[0];   // (8192,1,28,28)
    const float* params = (const float*)d_in[1];   // (4,4,3)
    const float* w      = (const float*)d_in[2];   // (10,784)
    const float* bias   = (const float*)d_in[3];   // (10)
    float* out = (float*)d_out;                    // (8192,10)

    _Float16* w2   = (_Float16*)d_ws;                                   // 50176 B
    float*    bp   = (float*)((char*)d_ws + (size_t)W2_ELEMS * 2);      // 16 floats

    quanv_precompute<<<1, 256, 0, stream>>>(params, w, bias, w2, bp);
    quanv_main<<<64, 256, 0, stream>>>(x, w2, bp, out);
}
